// TreeMultiRandom_55190329753603
// MI455X (gfx1250) — compile-verified, matrix operand order checked by reading, not run on hardware
//
#include <hip/hip_runtime.h>
#include <stdint.h>

// ---------------------------------------------------------------------------
// KNN channel fill (TreeMultiRandom) for MI455X / gfx1250, wave32 + WMMA.
//
// d2(q,p) = |p|^2 + |q|^2 - 2 p.q ; the p.q term for a 16x16 (cand x query)
// tile is one v_wmma_f32_16x16x32_f16 (K padded 2->32 with zeros).
// Coordinates i/256 are exact in f16 and every intermediate product/sum is
// exact in the f32 accumulator, so d2 matches the reference bit-for-bit.
//
// Inner loop: 2 candidate tiles per iteration (2 WMMAs back-to-back to hide
// the WMMA->VALU coexec hazard), per-tile min8 pre-filter so the branchless
// top-4 insertion block is skipped via one cmpx/cbranch in steady state.
// ---------------------------------------------------------------------------

#define SZ     256
#define N_PIX  65536
#define NCH    8
#define CAP    65792            // 65536 + 8*32 pad, multiple of 32

#define WS_BLOCKCNT 0
#define WS_WRITEPOS 2048
#define WS_CHBASE   4096
#define WS_CHTILES  4104
#define WS_YX       4352
#define WS_NORM     (WS_YX + CAP)
#define WS_VAL      (WS_YX + 2 * CAP)

typedef _Float16 v16h __attribute__((ext_vector_type(16)));
typedef float    v8f  __attribute__((ext_vector_type(8)));

__device__ __forceinline__ uint32_t pack_half2(float y, float x) {
    _Float16 hy = (_Float16)y, hx = (_Float16)x;
    return (uint32_t)__builtin_bit_cast(unsigned short, hy) |
           ((uint32_t)__builtin_bit_cast(unsigned short, hx) << 16);
}

// keep (min key, tie -> min idx) in (ka,ia)
__device__ __forceinline__ void cswap(float& ka, int& ia, float& kb, int& ib) {
    bool sw = (kb < ka) || (kb == ka && ib < ia);
    float tk = sw ? kb : ka; int ti = sw ? ib : ia;
    kb = sw ? ka : kb;       ib = sw ? ia : ib;
    ka = tk;                 ia = ti;
}

// branchless sorted insert of (key,idx) into ascending (k0..k3,i0..i3)
__device__ __forceinline__ void insert4(float key, int idx,
                                        float& k0, float& k1, float& k2, float& k3,
                                        int& i0, int& i1, int& i2, int& i3) {
    bool c3 = key < k3, c2 = key < k2, c1 = key < k1, c0 = key < k0;
    float nk3 = c3 ? (c2 ? k2 : key) : k3;  int ni3 = c3 ? (c2 ? i2 : idx) : i3;
    float nk2 = c2 ? (c1 ? k1 : key) : k2;  int ni2 = c2 ? (c1 ? i1 : idx) : i2;
    float nk1 = c1 ? (c0 ? k0 : key) : k1;  int ni1 = c1 ? (c0 ? i0 : idx) : i1;
    float nk0 = c0 ? key : k0;              int ni0 = c0 ? idx : i0;
    k0 = nk0; k1 = nk1; k2 = nk2; k3 = nk3;
    i0 = ni0; i1 = ni1; i2 = ni2; i3 = ni3;
}

// ---- pass 1: per-(channel,row) counts -------------------------------------
__global__ void tmr_count(const int* __restrict__ lut, int* __restrict__ blockCnt) {
    __shared__ int cnt[NCH];
    int t = threadIdx.x, b = blockIdx.x;
    if (t < NCH) cnt[t] = 0;
    __syncthreads();
    int c = lut[b * 256 + t] - 1;
    if (c >= 0 && c < NCH) atomicAdd(&cnt[c], 1);
    __syncthreads();
    if (t < NCH) blockCnt[t * 256 + b] = cnt[t];
}

// ---- pass 2: serial scan + sentinel padding (tiny) ------------------------
__global__ void tmr_scan(int* __restrict__ ws) {
    int*      blockCnt = ws + WS_BLOCKCNT;
    int*      writePos = ws + WS_WRITEPOS;
    int*      chBase   = ws + WS_CHBASE;
    int*      chTiles  = ws + WS_CHTILES;
    uint32_t* yx  = (uint32_t*)(ws + WS_YX);
    float*    nrm = (float*)(ws + WS_NORM);
    float*    val = (float*)(ws + WS_VAL);
    int base = 0;
    for (int c = 0; c < NCH; ++c) {
        int run = base;
        for (int b = 0; b < 256; ++b) { writePos[c * 256 + b] = run; run += blockCnt[c * 256 + b]; }
        int total  = run - base;
        int padded = (total + 31) & ~31;      // multiple of 32 -> even tile count
        chBase[c]  = base;
        chTiles[c] = padded >> 4;
        for (int p = base + total; p < base + padded; ++p) {
            yx[p] = 0u; nrm[p] = 3.0e8f; val[p] = 0.0f;   // sentinel: never wins
        }
        base += padded;
    }
}

// ---- pass 3: stable (index-ordered) compaction, one row per block ---------
__global__ void tmr_compact(const int* __restrict__ lut, const float* __restrict__ img,
                            int* __restrict__ ws) {
    __shared__ int   slut[256];
    __shared__ float simg[256];
    int t = threadIdx.x, b = blockIdx.x;
    slut[t] = lut[b * 256 + t];
    simg[t] = img[b * 256 + t];
    __syncthreads();
    if (t == 0) {
        int*      writePos = ws + WS_WRITEPOS;
        uint32_t* yx  = (uint32_t*)(ws + WS_YX);
        float*    nrm = (float*)(ws + WS_NORM);
        float*    val = (float*)(ws + WS_VAL);
        int cur[NCH];
        #pragma unroll
        for (int c = 0; c < NCH; ++c) cur[c] = writePos[c * 256 + b];
        float py = (float)b * (1.0f / SZ);
        for (int j = 0; j < 256; ++j) {
            int c = slut[j] - 1;
            if (c < 0 || c >= NCH) continue;
            float px = (float)j * (1.0f / SZ);
            int p = cur[c]++;
            yx[p]  = pack_half2(py, px);
            nrm[p] = py * py + px * px;
            val[p] = simg[j];
        }
    }
}

// ---- pass 4: WMMA brute-force KNN + distance-weighted fill ----------------
// grid (512, 8), block 256 (8 waves). Each wave: 16 queries of one channel.
__global__ __launch_bounds__(256)
void tmr_knn(const float* __restrict__ img, const int* __restrict__ lut,
             const int* __restrict__ ws, float* __restrict__ out) {
    const int ch   = blockIdx.y;
    const int wave = threadIdx.x >> 5;
    const int lane = threadIdx.x & 31;
    const int n    = lane & 15;          // query slot (N dim)
    const int hi   = lane >> 4;          // half: candidates m+0..7 vs m+8..15
    const int q    = (blockIdx.x * 8 + wave) * 16 + n;

    const int cbase  = ws[WS_CHBASE + ch];
    const int ntiles = ws[WS_CHTILES + ch];     // even
    const uint32_t* yx  = (const uint32_t*)(ws + WS_YX);
    const float*    nrm = (const float*)(ws + WS_NORM);
    const float*    val = (const float*)(ws + WS_VAL);

    const float qy    = (float)(q >> 8) * (1.0f / SZ);
    const float qx    = (float)(q & 255) * (1.0f / SZ);
    const float qnorm = qy * qy + qx * qx;

    union V16 { v16h h; uint32_t u[8]; };
    V16 B, A0, A1;
    #pragma unroll
    for (int i = 0; i < 8; ++i) { B.u[i] = 0u; A0.u[i] = 0u; A1.u[i] = 0u; }
    // B (32x16 f16): lanes 0-15 hold col N=lane, VGPR0 = K0,K1 = (qy,qx)
    B.u[0] = (lane < 16) ? pack_half2(qy, qx) : 0u;

    float k0 = 1e30f, k1 = 1e30f, k2 = 1e30f, k3 = 1e30f;   // key = |p|^2 - 2 p.q
    int   i0 = 0,     i1 = 0,     i2 = 0,     i3 = 0;

    for (int t = 0; t < ntiles; t += 2) {
        const int mBase0 = t << 4;
        const int mBase1 = mBase0 + 16;
        // A (16x32 f16): lanes 0-15 hold row M=lane, VGPR0 = K0,K1 = (py,px)
        uint32_t cyx0 = yx[cbase + mBase0 + n];
        uint32_t cyx1 = yx[cbase + mBase1 + n];
        A0.u[0] = (lane < 16) ? cyx0 : 0u;
        A1.u[0] = (lane < 16) ? cyx1 : 0u;

        v8f cz = {};
        v8f dot0 = __builtin_amdgcn_wmma_f32_16x16x32_f16(
            false, A0.h, false, B.h, (short)0, cz, false, false);
        v8f dot1 = __builtin_amdgcn_wmma_f32_16x16x32_f16(
            false, A1.h, false, B.h, (short)0, cz, false, false);

        // 8 candidate norms per half-tile (broadcast 32B, L0/L2-resident)
        const float4* np0 = reinterpret_cast<const float4*>(nrm + cbase + mBase0 + hi * 8);
        const float4* np1 = reinterpret_cast<const float4*>(nrm + cbase + mBase1 + hi * 8);
        float4 nA0 = np0[0], nB0 = np0[1];
        float4 nA1 = np1[0], nB1 = np1[1];

        {   // ---- tile 0 ----
            float nr[8] = {nA0.x, nA0.y, nA0.z, nA0.w, nB0.x, nB0.y, nB0.z, nB0.w};
            float key[8];
            #pragma unroll
            for (int r = 0; r < 8; ++r) key[r] = fmaf(-2.0f, dot0[r], nr[r]);
            float m8 = fminf(fminf(fminf(key[0], key[1]), fminf(key[2], key[3])),
                             fminf(fminf(key[4], key[5]), fminf(key[6], key[7])));
            if (m8 < k3) {                     // one divergent branch per tile
                const int ib = mBase0 + hi * 8;
                #pragma unroll
                for (int r = 0; r < 8; ++r)
                    insert4(key[r], ib + r, k0, k1, k2, k3, i0, i1, i2, i3);
            }
        }
        {   // ---- tile 1 ----
            float nr[8] = {nA1.x, nA1.y, nA1.z, nA1.w, nB1.x, nB1.y, nB1.z, nB1.w};
            float key[8];
            #pragma unroll
            for (int r = 0; r < 8; ++r) key[r] = fmaf(-2.0f, dot1[r], nr[r]);
            float m8 = fminf(fminf(fminf(key[0], key[1]), fminf(key[2], key[3])),
                             fminf(fminf(key[4], key[5]), fminf(key[6], key[7])));
            if (m8 < k3) {
                const int ib = mBase1 + hi * 8;
                #pragma unroll
                for (int r = 0; r < 8; ++r)
                    insert4(key[r], ib + r, k0, k1, k2, k3, i0, i1, i2, i3);
            }
        }
    }

    // merge the two sorted top-4 lists of lane and lane^16 (Batcher split+sort)
    float b0 = __shfl_xor(k0, 16, 32), b1 = __shfl_xor(k1, 16, 32),
          b2 = __shfl_xor(k2, 16, 32), b3 = __shfl_xor(k3, 16, 32);
    int   j0 = __shfl_xor(i0, 16, 32), j1 = __shfl_xor(i1, 16, 32),
          j2 = __shfl_xor(i2, 16, 32), j3 = __shfl_xor(i3, 16, 32);
    cswap(k0, i0, b3, j3); cswap(k1, i1, b2, j2);
    cswap(k2, i2, b1, j1); cswap(k3, i3, b0, j0);
    cswap(k0, i0, k2, i2); cswap(k1, i1, k3, i3);
    cswap(k0, i0, k1, i1); cswap(k2, i2, k3, i3);
    cswap(k1, i1, k2, i2);

    // weighted fill: sum(v * dist) / sum(dist)  (reference weights by distance)
    float d0 = sqrtf(fmaxf(k0 + qnorm, 0.0f));
    float d1 = sqrtf(fmaxf(k1 + qnorm, 0.0f));
    float d2 = sqrtf(fmaxf(k2 + qnorm, 0.0f));
    float d3 = sqrtf(fmaxf(k3 + qnorm, 0.0f));
    float v0 = val[cbase + i0], v1 = val[cbase + i1];
    float v2 = val[cbase + i2], v3 = val[cbase + i3];
    float numr = ((v0 * d0 + v1 * d1) + v2 * d2) + v3 * d3;
    float den  = ((d0 + d1) + d2) + d3;
    float res  = numr / den;

    if (hi == 0) {
        int m = lut[q];
        out[ch * N_PIX + q] = (m == ch + 1) ? img[q] : res;
    }
}

// ---------------------------------------------------------------------------
extern "C" void kernel_launch(void* const* d_in, const int* in_sizes, int n_in,
                              void* d_out, int out_size, void* d_ws, size_t ws_size,
                              hipStream_t stream) {
    (void)in_sizes; (void)n_in; (void)out_size; (void)ws_size;
    const float* img = (const float*)d_in[0];     // coded [1,1,256,256] f32
    const int*   lut = (const int*)d_in[1];       // lookup_table [256,256] i32
    float*       out = (float*)d_out;             // [1,8,256,256] f32
    int*         ws  = (int*)d_ws;

    tmr_count  <<<256, 256, 0, stream>>>(lut, ws + WS_BLOCKCNT);
    tmr_scan   <<<1,   1,   0, stream>>>(ws);
    tmr_compact<<<256, 256, 0, stream>>>(lut, img, ws);
    tmr_knn    <<<dim3(512, NCH), 256, 0, stream>>>(img, lut, ws, out);
}
